// TransformerBlock_81836306858030
// MI455X (gfx1250) — compile-verified, matrix operand order checked by reading, not run on hardware
//
#include <hip/hip_runtime.h>

// ---------------------------------------------------------------------------
// Transformer block (pre-LN MHA + FFN) for MI455X (gfx1250), bf16 WMMA path.
// B=4, S=2048, D=1024, H=16, Dh=64, FF=4096.
// v3: fixes async-builtin pointer types (v4i AS1 / AS3). Pre-transposed bf16
// weights, V stored head-transposed for the attention PV stage, and
// double-buffered GEMM tiles staged with GLOBAL_LOAD_ASYNC_TO_LDS_B128.
// ---------------------------------------------------------------------------

#define TB 4
#define TS 2048
#define TD 1024
#define TH 16
#define TDH 64
#define TFF 4096
#define TM (TB * TS)   // 8192 rows

typedef __attribute__((ext_vector_type(16))) __bf16 v16bf;
typedef __attribute__((ext_vector_type(8)))  __bf16 bf16x8;
typedef __attribute__((ext_vector_type(8)))  float  v8f;
typedef __attribute__((ext_vector_type(4)))  int    int4v;

__device__ __forceinline__ v8f zero8() {
  v8f z;
#pragma unroll
  for (int i = 0; i < 8; ++i) z[i] = 0.0f;
  return z;
}

__device__ __forceinline__ v8f wmma_bf16(v16bf a, v16bf b, v8f c) {
  return __builtin_amdgcn_wmma_f32_16x16x32_bf16(false, a, false, b,
                                                 (short)0, c, false, false);
}

// 16-byte global->LDS copy; async DMA (ASYNCcnt) when the builtin exists.
__device__ __forceinline__ void copy16_g2l(__bf16* lds, const __bf16* g) {
#if defined(__gfx1250__) && __has_builtin(__builtin_amdgcn_global_load_async_to_lds_b128)
  __builtin_amdgcn_global_load_async_to_lds_b128(
      (__attribute__((address_space(1))) int4v*)(uintptr_t)g,
      (__attribute__((address_space(3))) int4v*)(unsigned int)(uintptr_t)lds,
      0, 0);
#else
  *reinterpret_cast<bf16x8*>(lds) = *reinterpret_cast<const bf16x8*>(g);
#endif
}

__device__ __forceinline__ void wait_g2l() {
#if defined(__gfx1250__) && __has_builtin(__builtin_amdgcn_s_wait_asynccnt)
  __builtin_amdgcn_s_wait_asynccnt(0);
#endif
}

// Load one 16x32 bf16 fragment (A layout; also used for B^T tiles stored
// row-major as [N][K]) from LDS.
//   lanes 0-15 : row = lane,    K = {kb+0..7, kb+16..23}
//   lanes 16-31: row = lane-16, K = {kb+8..15, kb+24..31}
__device__ __forceinline__ v16bf load_frag(const __bf16* lds, int row_base,
                                           int stride, int k_base) {
  int lane = threadIdx.x & 31;
  int r = lane & 15;
  int half = lane >> 4;
  const __bf16* p = lds + (row_base + r) * stride + k_base + half * 8;
  bf16x8 lo = *reinterpret_cast<const bf16x8*>(p);
  bf16x8 hi = *reinterpret_cast<const bf16x8*>(p + 16);
  v16bf f;
#pragma unroll
  for (int i = 0; i < 8; ++i) { f[i] = lo[i]; f[8 + i] = hi[i]; }
  return f;
}

// ---------------------------------------------------------------------------
// LayerNorm: one block per row of 1024, writes bf16.
// ---------------------------------------------------------------------------
__device__ __forceinline__ float block_sum(float v, float* red) {
#pragma unroll
  for (int off = 16; off; off >>= 1) v += __shfl_xor(v, off, 32);
  int wave = threadIdx.x >> 5;
  if ((threadIdx.x & 31) == 0) red[wave] = v;
  __syncthreads();
  float t = 0.0f;
  if (threadIdx.x < 8) {
    t = red[threadIdx.x];
#pragma unroll
    for (int off = 4; off; off >>= 1) t += __shfl_xor(t, off, 8);
    if (threadIdx.x == 0) red[0] = t;
  }
  __syncthreads();
  float r = red[0];
  __syncthreads();
  return r;
}

__global__ __launch_bounds__(256) void ln_kernel(const float* __restrict__ x,
                                                 const float* __restrict__ g,
                                                 const float* __restrict__ b,
                                                 __bf16* __restrict__ out) {
  __shared__ float red[8];
  size_t row = blockIdx.x;
  const float* xr = x + row * TD;
  float v[4];
  float s = 0.0f;
#pragma unroll
  for (int i = 0; i < 4; ++i) { v[i] = xr[threadIdx.x + i * 256]; s += v[i]; }
  float mean = block_sum(s, red) * (1.0f / TD);
  float sq = 0.0f;
#pragma unroll
  for (int i = 0; i < 4; ++i) { float d = v[i] - mean; sq += d * d; }
  float var = block_sum(sq, red) * (1.0f / TD);
  float rs = rsqrtf(var + 1e-5f);
  __bf16* orow = out + row * TD;
#pragma unroll
  for (int i = 0; i < 4; ++i) {
    int c = threadIdx.x + i * 256;
    orow[c] = (__bf16)((v[i] - mean) * rs * g[c] + b[c]);
  }
}

// ---------------------------------------------------------------------------
// Fused fp32 -> bf16 convert + transpose: src [K][N] fp32 -> dst [N][K] bf16.
// 32x32 tiles via padded LDS; both global accesses coalesced.
// ---------------------------------------------------------------------------
__global__ __launch_bounds__(256) void cvt_t_bf16(const float* __restrict__ src,
                                                  __bf16* __restrict__ dst,
                                                  int K, int N) {
  __shared__ float t[32][33];
  int k0 = blockIdx.y * 32, n0 = blockIdx.x * 32;
  int tx = threadIdx.x & 31, ty = threadIdx.x >> 5;   // 32 x 8
#pragma unroll
  for (int i = 0; i < 4; ++i) {
    int k = ty + i * 8;
    t[k][tx] = src[(size_t)(k0 + k) * N + n0 + tx];
  }
  __syncthreads();
#pragma unroll
  for (int i = 0; i < 4; ++i) {
    int n = ty + i * 8;
    dst[(size_t)(n0 + n) * K + k0 + tx] = (__bf16)t[tx][n];
  }
}

// ---------------------------------------------------------------------------
// WMMA GEMM:  C[M,N] = A[M,K](bf16) * Wt^T (Wt is [N][K] bf16) + epilogue.
// Block tile 128x128, BK=32, 256 threads = 8 waves (2x4); each wave owns a
// 64x32 sub-tile = 4x2 wmma accumulators.  Double-buffered LDS tiles staged
// with async global->LDS b128 DMA.
// MODE 0: outB bf16 in (B,H,S,Dh) head layout, +bias        (Q,K proj)
// MODE 4: outB bf16 in (B,H,Dh,S) transposed head layout    (V proj)
// MODE 1: outF = resid + C + bias (fp32)                    (O proj / FFN2)
// MODE 2: outB = gelu(C + bias) (bf16)                      (FFN1)
// ---------------------------------------------------------------------------
template <int MODE>
__global__ __launch_bounds__(256) void gemm_wmma(
    const __bf16* __restrict__ A, const __bf16* __restrict__ Wt,
    const float* __restrict__ bias, const float* __restrict__ resid,
    float* __restrict__ outF, __bf16* __restrict__ outB, int M, int N, int K) {
  __shared__ __bf16 sA[2][128 * 32];   // [m][k]
  __shared__ __bf16 sB[2][128 * 32];   // [n][k]
  int tid = threadIdx.x;
  int lane = tid & 31;
  int wave = tid >> 5;
  int wm = wave >> 2;   // 0..1
  int wn = wave & 3;    // 0..3
  int m0 = blockIdx.y * 128;
  int n0 = blockIdx.x * 128;

  v8f acc[4][2];
#pragma unroll
  for (int mt = 0; mt < 4; ++mt)
#pragma unroll
    for (int nt = 0; nt < 2; ++nt) acc[mt][nt] = zero8();

  int seg_row = tid >> 2;            // 0..63
  int seg_cs  = (tid & 3) * 8;       // 0/8/16/24

  auto stage = [&](int buf, int k0) {
    // A tile: 128x32, two 16B segments per thread.
    copy16_g2l(&sA[buf][seg_row * 32 + seg_cs],
               &A[(size_t)(m0 + seg_row) * K + k0 + seg_cs]);
    copy16_g2l(&sA[buf][(seg_row + 64) * 32 + seg_cs],
               &A[(size_t)(m0 + seg_row + 64) * K + k0 + seg_cs]);
    // B tile (Wt rows): 128x32, same pattern.
    copy16_g2l(&sB[buf][seg_row * 32 + seg_cs],
               &Wt[(size_t)(n0 + seg_row) * K + k0 + seg_cs]);
    copy16_g2l(&sB[buf][(seg_row + 64) * 32 + seg_cs],
               &Wt[(size_t)(n0 + seg_row + 64) * K + k0 + seg_cs]);
  };

  int nk = K >> 5;
  stage(0, 0);
  for (int t = 0; t < nk; ++t) {
    int buf = t & 1;
    wait_g2l();
    __syncthreads();                       // tile t visible everywhere
    if (t + 1 < nk) stage(buf ^ 1, (t + 1) << 5);

    const __bf16* cA = sA[buf];
    const __bf16* cB = sB[buf];
    v16bf bfr[2];
#pragma unroll
    for (int nt = 0; nt < 2; ++nt)
      bfr[nt] = load_frag(cB, wn * 32 + nt * 16, 32, 0);
#pragma unroll
    for (int mt = 0; mt < 4; ++mt) {
      v16bf af = load_frag(cA, wm * 64 + mt * 16, 32, 0);
#pragma unroll
      for (int nt = 0; nt < 2; ++nt)
        acc[mt][nt] = wmma_bf16(af, bfr[nt], acc[mt][nt]);
    }
    __syncthreads();                       // reads done before t+2 overwrite
  }

  // Epilogue. C layout: VGPR r -> row r (lanes 0-15) / r+8 (lanes 16-31),
  // column = lane&15 within the 16x16 tile.
  int r15 = lane & 15;
  int half = lane >> 4;
#pragma unroll
  for (int mt = 0; mt < 4; ++mt) {
#pragma unroll
    for (int nt = 0; nt < 2; ++nt) {
#pragma unroll
      for (int r = 0; r < 8; ++r) {
        int row = m0 + wm * 64 + mt * 16 + r + half * 8;
        int col = n0 + wn * 32 + nt * 16 + r15;
        float v = acc[mt][nt][r] + bias[col];
        if constexpr (MODE == 0) {
          int b = row >> 11;            // row / TS
          int s = row & (TS - 1);
          int h = col >> 6;             // col / TDH
          int d = col & (TDH - 1);
          outB[(((size_t)b * TH + h) * TS + s) * TDH + d] = (__bf16)v;
        } else if constexpr (MODE == 4) {
          int b = row >> 11;
          int s = row & (TS - 1);
          int h = col >> 6;
          int d = col & (TDH - 1);
          outB[(((size_t)b * TH + h) * TDH + d) * TS + s] = (__bf16)v;
        } else if constexpr (MODE == 1) {
          size_t idx = (size_t)row * N + col;
          outF[idx] = resid[idx] + v;
        } else if constexpr (MODE == 2) {
          float ge = 0.5f * v * (1.0f + erff(v * 0.70710678118654752f));
          outB[(size_t)row * N + col] = (__bf16)ge;
        }
      }
    }
  }
}

// ---------------------------------------------------------------------------
// Causal flash attention: one wave per (b,h,16-query tile).
// Q,K bf16 in (B*H, S, Dh); V bf16 in (B*H, Dh, S) (pre-transposed).
// Online softmax over 32-key chunks.
// QK^T: 2 chained wmma (K-dim = Dh = 64).  P*V: 4 wmma (N = 64).
// ---------------------------------------------------------------------------
__global__ __launch_bounds__(32) void attn_kernel(const __bf16* __restrict__ Q,
                                                  const __bf16* __restrict__ K,
                                                  const __bf16* __restrict__ Vt,
                                                  __bf16* __restrict__ ctx) {
  __shared__ __bf16 sQ[16 * 64];
  __shared__ __bf16 sK[32 * 64];    // [key][d]
  __shared__ __bf16 sVt[64 * 32];   // [d][key]
  __shared__ __bf16 sP[16 * 32];    // probs [q][key]

  int qt = blockIdx.x & (TS / 16 - 1);
  int bh = blockIdx.x / (TS / 16);
  int q0 = qt * 16;
  int lane = threadIdx.x;
  int r15 = lane & 15;
  int half = lane >> 4;

  const __bf16* Qb  = Q  + (size_t)bh * TS * TDH;
  const __bf16* Kb  = K  + (size_t)bh * TS * TDH;
  const __bf16* Vtb = Vt + (size_t)bh * TDH * TS;   // row d: TS keys

  // Load the 16x64 Q tile.
  {
    int r = lane >> 1;
    int c = (lane & 1) * 32;
#pragma unroll
    for (int i = 0; i < 4; ++i)
      *reinterpret_cast<bf16x8*>(&sQ[r * 64 + c + i * 8]) =
          *reinterpret_cast<const bf16x8*>(&Qb[(size_t)(q0 + r) * 64 + c + i * 8]);
  }
  __syncthreads();
  v16bf aQ0 = load_frag(sQ, 0, 64, 0);
  v16bf aQ1 = load_frag(sQ, 0, 64, 32);

  float mrow[8], lrow[8];
  v8f o[4];
#pragma unroll
  for (int r = 0; r < 8; ++r) { mrow[r] = -__builtin_inff(); lrow[r] = 0.0f; }
#pragma unroll
  for (int j = 0; j < 4; ++j) o[j] = zero8();

  int q_end = q0 + 15;
  for (int k0 = 0; k0 <= q_end; k0 += 32) {
    // Stage K chunk rows [key][d]: one key row per lane, 8x16B.
#pragma unroll
    for (int i = 0; i < 8; ++i)
      *reinterpret_cast<bf16x8*>(&sK[lane * 64 + i * 8]) =
          *reinterpret_cast<const bf16x8*>(&Kb[(size_t)(k0 + lane) * 64 + i * 8]);
    // Stage V^T chunk [d][key]: two d-rows per lane, contiguous copies.
#pragma unroll
    for (int dd = 0; dd < 2; ++dd) {
      int d = lane + dd * 32;
#pragma unroll
      for (int i = 0; i < 4; ++i)
        *reinterpret_cast<bf16x8*>(&sVt[d * 32 + i * 8]) =
            *reinterpret_cast<const bf16x8*>(&Vtb[(size_t)d * TS + k0 + i * 8]);
    }
    __syncthreads();

    // Scores for two 16-key subtiles (K-dim = 64 -> 2 chained wmma each).
    v8f s[2];
#pragma unroll
    for (int t = 0; t < 2; ++t) {
      v8f z = zero8();
      v16bf b0 = load_frag(sK, t * 16, 64, 0);
      v16bf b1 = load_frag(sK, t * 16, 64, 32);
      z = wmma_bf16(aQ0, b0, z);
      z = wmma_bf16(aQ1, b1, z);
      s[t] = z;
    }
    // Scale + causal mask.
#pragma unroll
    for (int t = 0; t < 2; ++t) {
      int col = k0 + t * 16 + r15;
#pragma unroll
      for (int r = 0; r < 8; ++r) {
        int row = q0 + r + half * 8;
        float v = s[t][r] * 0.125f;  // 1/sqrt(64)
        s[t][r] = (col <= row) ? v : -__builtin_inff();
      }
    }
    // Online softmax (row stats reduced across 16-lane groups).
    float alpha[8];
#pragma unroll
    for (int r = 0; r < 8; ++r) {
      float mx = fmaxf(s[0][r], s[1][r]);
#pragma unroll
      for (int off = 8; off; off >>= 1) mx = fmaxf(mx, __shfl_xor(mx, off, 16));
      float mnew = fmaxf(mrow[r], mx);
      alpha[r] = expf(mrow[r] - mnew);
      mrow[r] = mnew;
      float p0 = expf(s[0][r] - mnew);
      float p1 = expf(s[1][r] - mnew);
      s[0][r] = p0; s[1][r] = p1;
      float ls = p0 + p1;
#pragma unroll
      for (int off = 8; off; off >>= 1) ls += __shfl_xor(ls, off, 16);
      lrow[r] = lrow[r] * alpha[r] + ls;
    }
#pragma unroll
    for (int j = 0; j < 4; ++j)
#pragma unroll
      for (int r = 0; r < 8; ++r) o[j][r] *= alpha[r];

    // Probs -> LDS (C layout scatter) then reload as A fragment.
#pragma unroll
    for (int t = 0; t < 2; ++t)
#pragma unroll
      for (int r = 0; r < 8; ++r)
        sP[(r + half * 8) * 32 + t * 16 + r15] = (__bf16)s[t][r];
    __syncthreads();
    v16bf aP = load_frag(sP, 0, 32, 0);
#pragma unroll
    for (int j = 0; j < 4; ++j) {
      v16bf bV = load_frag(sVt, j * 16, 32, 0);
      o[j] = wmma_bf16(aP, bV, o[j]);
    }
    __syncthreads();
  }

  // Normalize and write ctx back in (B,S,D) layout as bf16.
  int b = bh / TH, h = bh % TH;
#pragma unroll
  for (int j = 0; j < 4; ++j)
#pragma unroll
    for (int r = 0; r < 8; ++r) {
      int row = q0 + r + half * 8;
      ctx[((size_t)(b * TS + row)) * TD + h * TDH + j * 16 + r15] =
          (__bf16)(o[j][r] / lrow[r]);
    }
}

// ---------------------------------------------------------------------------
// Launch sequence.
// ---------------------------------------------------------------------------
extern "C" void kernel_launch(void* const* d_in, const int* in_sizes, int n_in,
                              void* d_out, int out_size, void* d_ws, size_t ws_size,
                              hipStream_t stream) {
  const float* x     = (const float*)d_in[0];
  // d_in[1] = mask (causality handled analytically)
  const float* Wq    = (const float*)d_in[2];
  const float* bq    = (const float*)d_in[3];
  const float* Wk    = (const float*)d_in[4];
  const float* bk    = (const float*)d_in[5];
  const float* Wv    = (const float*)d_in[6];
  const float* bv    = (const float*)d_in[7];
  const float* Wo    = (const float*)d_in[8];
  const float* bo    = (const float*)d_in[9];
  const float* ln1_g = (const float*)d_in[10];
  const float* ln1_b = (const float*)d_in[11];
  const float* ln2_g = (const float*)d_in[12];
  const float* ln2_b = (const float*)d_in[13];
  const float* W1    = (const float*)d_in[14];
  const float* b1    = (const float*)d_in[15];
  const float* W2    = (const float*)d_in[16];
  const float* b2    = (const float*)d_in[17];
  float* out = (float*)d_out;

  char* ws = (char*)d_ws;
  size_t off = 0;
  auto alloc = [&](size_t bytes) -> void* {
    void* p = ws + off;
    off += (bytes + 255) & ~(size_t)255;
    return p;
  };
  __bf16* hln  = (__bf16*)alloc((size_t)TM * TD * 2);
  __bf16* WqT  = (__bf16*)alloc((size_t)TD * TD * 2);    // [N][K]
  __bf16* WkT  = (__bf16*)alloc((size_t)TD * TD * 2);
  __bf16* WvT  = (__bf16*)alloc((size_t)TD * TD * 2);
  __bf16* WoT  = (__bf16*)alloc((size_t)TD * TD * 2);
  __bf16* W1T  = (__bf16*)alloc((size_t)TFF * TD * 2);   // [4096][1024]
  __bf16* W2T  = (__bf16*)alloc((size_t)TD * TFF * 2);   // [1024][4096]
  __bf16* Qh   = (__bf16*)alloc((size_t)TM * TD * 2);    // (B,H,S,Dh)
  __bf16* Kh   = (__bf16*)alloc((size_t)TM * TD * 2);    // (B,H,S,Dh)
  __bf16* Vth  = (__bf16*)alloc((size_t)TM * TD * 2);    // (B,H,Dh,S)
  __bf16* ctx  = (__bf16*)alloc((size_t)TM * TD * 2);
  float*  x1   = (float*)alloc((size_t)TM * TD * 4);
  __bf16* h2   = (__bf16*)alloc((size_t)TM * TD * 2);
  __bf16* act  = (__bf16*)alloc((size_t)TM * TFF * 2);
  (void)ws_size; (void)in_sizes; (void)n_in; (void)out_size;

  // 1. LN1 -> bf16 h
  ln_kernel<<<TM, 256, 0, stream>>>(x, ln1_g, ln1_b, hln);

  // 2. Weight convert+transpose fp32[K][N] -> bf16[N][K]
  dim3 tDD(TD / 32, TD / 32);
  cvt_t_bf16<<<tDD, 256, 0, stream>>>(Wq, WqT, TD, TD);
  cvt_t_bf16<<<tDD, 256, 0, stream>>>(Wk, WkT, TD, TD);
  cvt_t_bf16<<<tDD, 256, 0, stream>>>(Wv, WvT, TD, TD);
  cvt_t_bf16<<<tDD, 256, 0, stream>>>(Wo, WoT, TD, TD);
  cvt_t_bf16<<<dim3(TFF / 32, TD / 32), 256, 0, stream>>>(W1, W1T, TD, TFF);
  cvt_t_bf16<<<dim3(TD / 32, TFF / 32), 256, 0, stream>>>(W2, W2T, TFF, TD);

  // 3. QKV projections (WMMA): Q,K head-major; V head-transposed
  dim3 gDD(TD / 128, TM / 128);
  gemm_wmma<0><<<gDD, 256, 0, stream>>>(hln, WqT, bq, nullptr, nullptr, Qh, TM, TD, TD);
  gemm_wmma<0><<<gDD, 256, 0, stream>>>(hln, WkT, bk, nullptr, nullptr, Kh, TM, TD, TD);
  gemm_wmma<4><<<gDD, 256, 0, stream>>>(hln, WvT, bv, nullptr, nullptr, Vth, TM, TD, TD);

  // 4. Causal flash attention (WMMA)
  attn_kernel<<<TB * TH * (TS / 16), 32, 0, stream>>>(Qh, Kh, Vth, ctx);

  // 5. O projection + residual -> x1 (fp32)
  gemm_wmma<1><<<gDD, 256, 0, stream>>>(ctx, WoT, bo, x, x1, nullptr, TM, TD, TD);

  // 6. LN2 -> bf16 h2
  ln_kernel<<<TM, 256, 0, stream>>>(x1, ln2_g, ln2_b, h2);

  // 7. FFN1 + exact GELU -> bf16 act
  dim3 gDF(TFF / 128, TM / 128);
  gemm_wmma<2><<<gDF, 256, 0, stream>>>(h2, W1T, b1, nullptr, nullptr, act, TM, TFF, TD);

  // 8. FFN2 + residual -> out (fp32)
  gemm_wmma<1><<<gDD, 256, 0, stream>>>(act, W2T, b2, x1, out, nullptr, TM, TD, TFF);
}
